// NodeAnomalyAwareModel_7103875908246
// MI455X (gfx1250) — matrix operand order repaired and checked by reading, not compiled, not run on hardware
//
#include <hip/hip_runtime.h>

#define IN_DIM 1433
#define KPAD   1440   // 45 * 32
#define NCHUNK 45
#define H64    64
#define A32    32
#define NC     7
#define NCOLS  96     // 64 (gcn) + 32 (sem)
#define MTILE  128
#define ASTR   40     // LDS row stride (ushorts) for A, padded to dodge bank conflicts
#define BSTR   40     // LDS row stride (ushorts) for B-transposed

typedef __attribute__((ext_vector_type(16))) __bf16 v16bf;
typedef __attribute__((ext_vector_type(8)))  float  v8f;

__device__ __forceinline__ unsigned short f2bf(float f) {
  unsigned int u = __float_as_uint(f);
  u += 0x7FFFu + ((u >> 16) & 1u);   // round-to-nearest-even
  return (unsigned short)(u >> 16);
}

// ---- K0: pack W_gcn[1433,64] | W_ps[1433,32] into bf16, per-K-chunk column-major:
//      Wpack[(chunk*96 + c)*32 + (k&31)] = W[(chunk*32 + k), c]   (zero-padded K)
__global__ void k_pack_w(const float* __restrict__ Wg, const float* __restrict__ Wp,
                         unsigned short* __restrict__ Wpack) {
  int t = blockIdx.x * blockDim.x + threadIdx.x;
  if (t >= KPAD * NCOLS) return;
  int k = t / NCOLS, c = t - k * NCOLS;
  float v = 0.f;
  if (k < IN_DIM) v = (c < H64) ? Wg[k * H64 + c] : Wp[k * A32 + (c - H64)];
  Wpack[(size_t)((k >> 5) * NCOLS + c) * 32 + (k & 31)] = f2bf(v);
}

// ---- degree: self-loop init, edge atomics, rsqrt
__global__ void k_deg_init(float* __restrict__ deg, int n) {
  int t = blockIdx.x * blockDim.x + threadIdx.x;
  if (t < n) deg[t] = 1.0f;
}
__global__ void k_deg_edges(const long long* __restrict__ ei, float* __restrict__ deg, int E) {
  int t = blockIdx.x * blockDim.x + threadIdx.x;
  if (t < E) atomicAdd(&deg[(int)ei[(size_t)E + t]], 1.0f);
}
__global__ void k_dis(const float* __restrict__ deg, float* __restrict__ dis, int n) {
  int t = blockIdx.x * blockDim.x + threadIdx.x;
  if (t < n) dis[t] = rsqrtf(deg[t]);
}

// ---- K3: fused bf16 WMMA GEMM: [N,1433] x [1433,96] -> xw[N,64] (no bias), zsem[N,32] (+b_ps)
__global__ __launch_bounds__(256)
void k_gemm(const float* __restrict__ x, const unsigned short* __restrict__ Wpack,
            const float* __restrict__ b_ps, float* __restrict__ xw,
            float* __restrict__ zsem, int n) {
  __shared__ __attribute__((aligned(16))) unsigned short Alds[MTILE * ASTR];
  __shared__ __attribute__((aligned(16))) unsigned short Blds[NCOLS * BSTR];

  const int tid  = threadIdx.x;
  const int wave = tid >> 5;
  const int lane = tid & 31;
  const int m0   = blockIdx.x * MTILE;

  v8f acc[6];
#pragma unroll
  for (int t = 0; t < 6; ++t)
#pragma unroll
    for (int j = 0; j < 8; ++j) acc[t][j] = 0.f;

  for (int kc = 0; kc < KPAD; kc += 32) {
    // ---- stage B tile via CDNA5 async copy (global -> LDS, no VGPR round-trip).
    //      Wpack chunk is column-major [96][32] bf16; each lane moves 16B into the
    //      padded transposed slot Blds[c*BSTR + k0].  Both sides 16B-aligned.
    {
      const unsigned chunkbase = (unsigned)(kc >> 5) * (NCOLS * 32);
      for (int idx = tid; idx < (NCOLS * 32) / 8; idx += 256) {   // 384 x B128
        int c  = idx >> 2;
        int k0 = (idx & 3) << 3;
        unsigned long long gsrc =
            (unsigned long long)(const void*)(Wpack + chunkbase + c * 32 + k0);
        unsigned ldst = (unsigned)(uintptr_t)(&Blds[c * BSTR + k0]);
        asm volatile("global_load_async_to_lds_b128 %0, %1, off"
                     :: "v"(ldst), "v"(gsrc) : "memory");
      }
    }
    // ---- stage A tile (128 x 32), f32 -> bf16 in flight; row + K<1433 bounds
    for (int idx = tid; idx < MTILE * 32; idx += 256) {
      int r = idx >> 5, k = idx & 31;
      int row = m0 + r, col = kc + k;
      float v = (row < n && col < IN_DIM) ? x[(size_t)row * IN_DIM + col] : 0.f;
      Alds[r * ASTR + k] = f2bf(v);
    }
    // ---- prefetch next K-chunk of A into L2/WGP$ while WMMAs run
    if (tid < MTILE) {
      int row = m0 + tid, col = kc + 32;
      if (row < n && col < IN_DIM)
        __builtin_prefetch(&x[(size_t)row * IN_DIM + col], 0, 1);
    }
    asm volatile("s_wait_asynccnt 0" ::: "memory");
    __syncthreads();

    // A fragment (16-bit 16x32 layout): lanes 0-15 rows, K {0..7,16..23}; lanes 16-31 K {8..15,24..31}
    union { v16bf v; uint4 q[2]; } fa;
    const int ar  = wave * 16 + (lane & 15);
    const int kb8 = (lane >> 4) << 3;                 // 0 or 8
    fa.q[0] = *(const uint4*)(&Alds[ar * ASTR + kb8]);
    fa.q[1] = *(const uint4*)(&Alds[ar * ASTR + kb8 + 16]);

    const int kb16 = (lane >> 4) << 4;                // 0 or 16
#pragma unroll
    for (int t = 0; t < 6; ++t) {
      union { v16bf v; uint4 q[2]; } fb;
      const int c = t * 16 + (lane & 15);
      fb.q[0] = *(const uint4*)(&Blds[c * BSTR + kb8]);
      fb.q[1] = *(const uint4*)(&Blds[c * BSTR + kb8 + 16]);
      acc[t] = __builtin_amdgcn_wmma_f32_16x16x32_bf16(
          false, fa.v, false, fb.v, (short)0, acc[t], false, false);
    }
    __syncthreads();
  }

  // C/D layout: VGPR j, lanes 0-15 -> M=j, lanes 16-31 -> M=j+8; col = tile*16 + lane%16
  const int rbase = m0 + wave * 16 + ((lane >> 4) << 3);
  const int cl = lane & 15;
  if (m0 + MTILE <= n) {
    // full tile: unconditional, coalesced stores
#pragma unroll
    for (int t = 0; t < 4; ++t) {
      int col = t * 16 + cl;
#pragma unroll
      for (int j = 0; j < 8; ++j)
        xw[(size_t)(rbase + j) * H64 + col] = acc[t][j];
    }
#pragma unroll
    for (int t = 4; t < 6; ++t) {
      int col = t * 16 + cl - H64;
      float b = b_ps[col];
#pragma unroll
      for (int j = 0; j < 8; ++j)
        zsem[(size_t)(rbase + j) * A32 + col] = acc[t][j] + b;
    }
  } else {
#pragma unroll
    for (int t = 0; t < 6; ++t) {
      int col = t * 16 + cl;
#pragma unroll
      for (int j = 0; j < 8; ++j) {
        int row = rbase + j;
        if (row < n) {
          float v = acc[t][j];
          if (col < H64) xw[(size_t)row * H64 + col] = v;
          else           zsem[(size_t)row * A32 + (col - H64)] = v + b_ps[col - H64];
        }
      }
    }
  }
}

// ---- K4: self-loop term: agg = xw * dis[n]^2
__global__ void k_agg_init(const float* __restrict__ xw, const float* __restrict__ dis,
                           float* __restrict__ agg, int n) {
  int t = blockIdx.x * blockDim.x + threadIdx.x;
  if (t < n * H64) {
    int nn = t >> 6;
    float d = dis[nn];
    agg[t] = xw[t] * d * d;
  }
}

// ---- K5: edge scatter: 16 lanes per edge, float4 gather + 4x global_atomic_add_f32
__global__ void k_scatter(const long long* __restrict__ ei, const float* __restrict__ dis,
                          const float* __restrict__ xw, float* __restrict__ agg, int E) {
  long long t = (long long)blockIdx.x * blockDim.x + threadIdx.x;
  if (t >= (long long)E * 16) return;
  int e = (int)(t >> 4), p = (int)(t & 15);
  int s = (int)ei[e];
  int d = (int)ei[(size_t)E + e];
  float w = dis[s] * dis[d];
  float4 v = ((const float4*)(xw + (size_t)s * H64))[p];
  float* dr = agg + (size_t)d * H64 + p * 4;
  atomicAdd(dr + 0, v.x * w);
  atomicAdd(dr + 1, v.y * w);
  atomicAdd(dr + 2, v.z * w);
  atomicAdd(dr + 3, v.w * w);
}

// ---- K6: h = relu(agg + b_gcn); z_topo = h @ W_pt + b_pt   (fp32, LDS-staged rows)
__global__ __launch_bounds__(256)
void k_ztopo(const float* __restrict__ agg, const float* __restrict__ b_gcn,
             const float* __restrict__ Wpt, const float* __restrict__ b_pt,
             float* __restrict__ ztopo, int n) {
  __shared__ float h[8][H64];
  int n0 = blockIdx.x * 8;
  int tid = threadIdx.x;
  for (int idx = tid; idx < 8 * H64; idx += 256) {
    int nl = idx >> 6, k = idx & 63;
    int node = n0 + nl;
    float v = 0.f;
    if (node < n) v = fmaxf(agg[(size_t)node * H64 + k] + b_gcn[k], 0.f);
    h[nl][k] = v;
  }
  __syncthreads();
  int nl = tid >> 5, c = tid & 31;
  int node = n0 + nl;
  if (node < n) {
    float s = b_pt[c];
#pragma unroll
    for (int k = 0; k < H64; ++k) s += h[nl][k] * Wpt[k * A32 + c];
    ztopo[(size_t)node * A32 + c] = s;
  }
}

// ---- K7: logits = z_topo @ W_cls + b_cls ; anomaly = ||z_topo - z_sem||
__global__ void k_head(const float* __restrict__ ztopo, const float* __restrict__ zsem,
                       const float* __restrict__ Wc, const float* __restrict__ bc,
                       float* __restrict__ logits, float* __restrict__ anom, int n) {
  int t = blockIdx.x * blockDim.x + threadIdx.x;
  if (t >= n) return;
  const float* zt = ztopo + (size_t)t * A32;
  const float* zs = zsem + (size_t)t * A32;
  float lg[NC];
#pragma unroll
  for (int c = 0; c < NC; ++c) lg[c] = bc[c];
  float s = 0.f;
#pragma unroll
  for (int k = 0; k < A32; ++k) {
    float v = zt[k];
    float d = v - zs[k];
    s += d * d;
#pragma unroll
    for (int c = 0; c < NC; ++c) lg[c] += v * Wc[k * NC + c];
  }
#pragma unroll
  for (int c = 0; c < NC; ++c) logits[(size_t)t * NC + c] = lg[c];
  anom[t] = sqrtf(s);
}

extern "C" void kernel_launch(void* const* d_in, const int* in_sizes, int n_in,
                              void* d_out, int out_size, void* d_ws, size_t ws_size,
                              hipStream_t stream) {
  const float*     x   = (const float*)d_in[0];
  const long long* ei  = (const long long*)d_in[1];
  const float*     Wg  = (const float*)d_in[2];
  const float*     bg  = (const float*)d_in[3];
  const float*     Wpt = (const float*)d_in[4];
  const float*     bpt = (const float*)d_in[5];
  const float*     Wps = (const float*)d_in[6];
  const float*     bps = (const float*)d_in[7];
  const float*     Wc  = (const float*)d_in[8];
  const float*     bc  = (const float*)d_in[9];

  const int n = in_sizes[0] / IN_DIM;     // 50000
  const int E = in_sizes[1] / 2;          // 1600000

  float* out    = (float*)d_out;
  float* logits = out;
  float* anom   = out + (size_t)n * NC;
  float* ztopo  = out + (size_t)n * (NC + 1);
  float* zsem   = out + (size_t)n * (NC + 1 + A32);

  auto rup = [](size_t o) { return (o + 255) & ~(size_t)255; };
  char* ws = (char*)d_ws;
  size_t o = 0;
  unsigned short* Wpack = (unsigned short*)(ws + o); o = rup(o + (size_t)KPAD * NCOLS * 2);
  float* deg = (float*)(ws + o); o = rup(o + (size_t)n * 4);
  float* dis = (float*)(ws + o); o = rup(o + (size_t)n * 4);
  float* xw  = (float*)(ws + o); o = rup(o + (size_t)n * H64 * 4);
  float* agg = (float*)(ws + o); o = rup(o + (size_t)n * H64 * 4);
  (void)ws_size; (void)n_in; (void)out_size;

  k_pack_w  <<<(KPAD * NCOLS + 255) / 256, 256, 0, stream>>>(Wg, Wps, Wpack);
  k_deg_init<<<(n + 255) / 256, 256, 0, stream>>>(deg, n);
  k_deg_edges<<<(E + 255) / 256, 256, 0, stream>>>(ei, deg, E);
  k_dis     <<<(n + 255) / 256, 256, 0, stream>>>(deg, dis, n);
  k_gemm    <<<(n + MTILE - 1) / MTILE, 256, 0, stream>>>(x, Wpack, bps, xw, zsem, n);
  k_agg_init<<<(n * H64 + 255) / 256, 256, 0, stream>>>(xw, dis, agg, n);
  {
    long long tot = (long long)E * 16;
    k_scatter<<<(unsigned int)((tot + 255) / 256), 256, 0, stream>>>(ei, dis, xw, agg, E);
  }
  k_ztopo   <<<(n + 7) / 8, 256, 0, stream>>>(agg, bg, Wpt, bpt, ztopo, n);
  k_head    <<<(n + 255) / 256, 256, 0, stream>>>(ztopo, zsem, Wc, bc, logits, anom, n);
}
